// AdaptGraphPooling_21809843929186
// MI455X (gfx1250) — matrix operand order repaired and checked by reading, not compiled, not run on hardware
//
#include <hip/hip_runtime.h>
#include <hip/hip_bf16.h>

typedef __attribute__((ext_vector_type(16))) _Float16 v16h;
typedef __attribute__((ext_vector_type(8)))  _Float16 v8h;
typedef __attribute__((ext_vector_type(8)))  float    v8f;

#define NB 8
#define NP 4096
#define NC 256
#define NM 1024
#define NK 16
#define ND 64
#define NOUT 259
#define TB_STRIDE 24    // halves per LDS row (48B -> 16B aligned, good bank spread)
#define GPE_STRIDE 264  // floats per gpe column (col-major: [k][c]); 264*4B = 16B aligned

// ---------------------------------------------------------------------------
// WMMA helpers (CDNA5 16x16x32 f16 -> f32)
// ---------------------------------------------------------------------------
__device__ __forceinline__ v8f wmma16(v16h a, v16h b, v8f c) {
  // 8 args: (neg_a, A, neg_b, B, c_mod, C, reuse_a, reuse_b)
  return __builtin_amdgcn_wmma_f32_16x16x32_f16(false, a, false, b, (short)0, c,
                                                false, false);
}

// A operand (16x32 f16 tile) from row-major memory.
// ISA 7.12.2 layout: lanes 0-15 (g=0): elems[0..7]=K0..7,  elems[8..15]=K16..23
//                    lanes16-31 (g=1): elems[0..7]=K8..15, elems[8..15]=K24..31
__device__ __forceinline__ v16h load_a(const _Float16* rowp, int g) {
  const v8h* p0 = (const v8h*)(rowp + 8 * g);
  const v8h* p1 = (const v8h*)(rowp + 16 + 8 * g);
  v8h lo = *p0, hi = *p1;
  v16h a;
#pragma unroll
  for (int i = 0; i < 8; ++i) { a[i] = lo[i]; a[8 + i] = hi[i]; }
  return a;
}

// B operand (32x16 f16): lane l holds column N=l%16; colp points at row 8g, column N.
// elems[0..7]=K8g..8g+7, elems[8..15]=K16+8g..16+8g+7 (rows TB_STRIDE apart).
__device__ __forceinline__ v16h load_b(const _Float16* colp) {
  v16h b;
#pragma unroll
  for (int i = 0; i < 8; ++i) { b[i] = colp[i * TB_STRIDE]; b[8 + i] = colp[(16 + i) * TB_STRIDE]; }
  return b;
}

// ---------------------------------------------------------------------------
// Cross-lane reductions via immediate-pattern ds_swizzle (no address VALU).
// offset = (xor_mask<<10) | and_mask(0x1f), group-of-32 mode.
// ---------------------------------------------------------------------------
template <int M>
__device__ __forceinline__ float sxorf(float x) {
  return __int_as_float(
      __builtin_amdgcn_ds_swizzle(__float_as_int(x), (M << 10) | 0x1f));
}
template <int M>
__device__ __forceinline__ int sxori(int x) {
  return __builtin_amdgcn_ds_swizzle(x, (M << 10) | 0x1f);
}
__device__ __forceinline__ float hredmax16(float x) {
  x = fmaxf(x, sxorf<1>(x));
  x = fmaxf(x, sxorf<2>(x));
  x = fmaxf(x, sxorf<4>(x));
  x = fmaxf(x, sxorf<8>(x));
  return x;
}
__device__ __forceinline__ float hredsum16(float x) {
  x += sxorf<1>(x);
  x += sxorf<2>(x);
  x += sxorf<4>(x);
  x += sxorf<8>(x);
  return x;
}

// ---------------------------------------------------------------------------
// Kernel 1: farthest point sampling. 1 block / batch, 1024 threads (32 waves).
// xyz in dynamic LDS, per-thread running min-dist (4 points each).
// ---------------------------------------------------------------------------
__global__ void fps_kernel(const float* __restrict__ vertices,
                           int* __restrict__ fps_idx) {
  extern __shared__ __align__(16) float fsm[];  // x[NP], y[NP], z[NP]
  float* sx = fsm; float* sy = fsm + NP; float* sz = fsm + 2 * NP;
  __shared__ float rmax[32];
  __shared__ int   rarg[32];
  __shared__ int   sfar;
  const int b = blockIdx.x, tid = threadIdx.x;
  const float* vb = vertices + (size_t)b * 3 * NP;
  for (int i = tid; i < NP; i += 1024) {
    sx[i] = vb[i]; sy[i] = vb[NP + i]; sz[i] = vb[2 * NP + i];
  }
  float dloc[4];
#pragma unroll
  for (int j = 0; j < 4; ++j) dloc[j] = 1e10f;
  int far = 0;
  __syncthreads();
  for (int it = 0; it < NM; ++it) {
    if (tid == 0) fps_idx[b * NM + it] = far;
    const float cx = sx[far], cy = sy[far], cz = sz[far];
    float bestd = -1.f; int besti = 0;
#pragma unroll
    for (int j = 0; j < 4; ++j) {
      const int i = tid + j * 1024;
      float dx = sx[i] - cx, dy = sy[i] - cy, dz = sz[i] - cz;
      float d = dx * dx + dy * dy + dz * dz;
      float dm = fminf(dloc[j], d);
      dloc[j] = dm;
      if (dm > bestd) { bestd = dm; besti = i; }  // strict > keeps lowest index
    }
    // wave32 argmax reduction (prefer lower index on ties) via ds_swizzle
    {
      float od; int oi;
      od = sxorf<16>(bestd); oi = sxori<16>(besti);
      if (od > bestd || (od == bestd && oi < besti)) { bestd = od; besti = oi; }
      od = sxorf<8>(bestd); oi = sxori<8>(besti);
      if (od > bestd || (od == bestd && oi < besti)) { bestd = od; besti = oi; }
      od = sxorf<4>(bestd); oi = sxori<4>(besti);
      if (od > bestd || (od == bestd && oi < besti)) { bestd = od; besti = oi; }
      od = sxorf<2>(bestd); oi = sxori<2>(besti);
      if (od > bestd || (od == bestd && oi < besti)) { bestd = od; besti = oi; }
      od = sxorf<1>(bestd); oi = sxori<1>(besti);
      if (od > bestd || (od == bestd && oi < besti)) { bestd = od; besti = oi; }
    }
    if ((tid & 31) == 0) { rmax[tid >> 5] = bestd; rarg[tid >> 5] = besti; }
    __syncthreads();
    if (tid == 0) {
      float bd = rmax[0]; int bi = rarg[0];
      for (int w = 1; w < 32; ++w)
        if (rmax[w] > bd || (rmax[w] == bd && rarg[w] < bi)) { bd = rmax[w]; bi = rarg[w]; }
      sfar = bi;
    }
    __syncthreads();
    far = sfar;
  }
}

// ---------------------------------------------------------------------------
// Kernel 2: 16-NN per key point. 1 thread / query, xyz tiled through LDS,
// branch-free sorted insertion (fully unrolled -> no scratch spills).
// ---------------------------------------------------------------------------
__global__ void knn_kernel(const float* __restrict__ vertices,
                           const int* __restrict__ fps_idx,
                           int* __restrict__ knn_idx) {
  extern __shared__ __align__(16) float ksm[];  // 3 * 1024 floats
  const int q = blockIdx.x * 256 + threadIdx.x;   // block spans a single batch
  const int b = q >> 10, m = q & (NM - 1);
  const float* vb = vertices + (size_t)b * 3 * NP;
  const int kidx = fps_idx[b * NM + m];
  const float qx = vb[kidx], qy = vb[NP + kidx], qz = vb[2 * NP + kidx];
  float bd[NK]; int bi[NK];
#pragma unroll
  for (int j = 0; j < NK; ++j) { bd[j] = 3.0e38f; bi[j] = 0; }
  for (int t = 0; t < 4; ++t) {
    __syncthreads();
    for (int i = threadIdx.x; i < 1024; i += 256) {
      const int n = t * 1024 + i;
      ksm[i] = vb[n]; ksm[1024 + i] = vb[NP + n]; ksm[2048 + i] = vb[2 * NP + n];
    }
    __syncthreads();
    for (int i = 0; i < 1024; ++i) {
      float dx = ksm[i] - qx, dy = ksm[1024 + i] - qy, dz = ksm[2048 + i] - qz;
      float d = dx * dx + dy * dy + dz * dz;
      if (d < bd[NK - 1]) {
        float cd = d; int ci = t * 1024 + i;
#pragma unroll
        for (int j = 0; j < NK; ++j) {
          if (cd < bd[j]) {
            float td = bd[j]; int ti = bi[j];
            bd[j] = cd; bi[j] = ci; cd = td; ci = ti;
          }
        }
      }
    }
  }
#pragma unroll
  for (int j = 0; j < NK; ++j) knn_idx[(size_t)q * NK + j] = bi[j];
}

// ---------------------------------------------------------------------------
// Kernel 3: weight prep — f32->f16 weights, aw2 row reorder (features first,
// xyz rows 256..258, zero pad to 272), folded BN scale/shift.
// ---------------------------------------------------------------------------
__global__ void prep_kernel(const float* __restrict__ pw2, const float* __restrict__ aw1,
                            const float* __restrict__ aw2, const float* __restrict__ ab2,
                            const float* __restrict__ bn1_g, const float* __restrict__ bn1_b,
                            const float* __restrict__ bn1_m, const float* __restrict__ bn1_v,
                            const float* __restrict__ bn2_g, const float* __restrict__ bn2_b,
                            const float* __restrict__ bn2_m, const float* __restrict__ bn2_v,
                            _Float16* __restrict__ pw2h, _Float16* __restrict__ aw1h,
                            _Float16* __restrict__ aw2rh, float* __restrict__ ab2r,
                            float* __restrict__ sc1, float* __restrict__ sh1,
                            float* __restrict__ sc2, float* __restrict__ sh2) {
  const int tid = blockIdx.x * blockDim.x + threadIdx.x;
  const int nt = gridDim.x * blockDim.x;
  for (int i = tid; i < NC * ND; i += nt) pw2h[i] = (_Float16)pw2[i];
  for (int i = tid; i < ND * NC; i += nt) aw1h[i] = (_Float16)aw1[i];
  for (int i = tid; i < 272 * ND; i += nt) {
    const int r = i >> 6, c = i & 63;
    float v = (r < 256) ? aw2[(size_t)(3 + r) * ND + c]
            : (r < 259) ? aw2[(size_t)(r - 256) * ND + c] : 0.f;
    aw2rh[i] = (_Float16)v;
  }
  for (int i = tid; i < 272; i += nt)
    ab2r[i] = (i < 256) ? ab2[3 + i] : (i < 259) ? ab2[i - 256] : 0.f;
  for (int i = tid; i < ND; i += nt) {
    float s1 = bn1_g[i] * rsqrtf(bn1_v[i] + 1e-5f);
    sc1[i] = s1; sh1[i] = bn1_b[i] - bn1_m[i] * s1;
    float s2 = bn2_g[i] * rsqrtf(bn2_v[i] + 1e-5f);
    sc2[i] = s2; sh2[i] = bn2_b[i] - bn2_m[i] * s2;
  }
}

// ---------------------------------------------------------------------------
// Kernel 4: fused gather + 3 WMMA GEMMs + softmax + weighted pooling.
// One wave per (b,m) group of 16 neighbors, 4 waves / block.
// ---------------------------------------------------------------------------
struct GroupLds {
  _Float16 tB[NC * TB_STRIDE];     // t = qk_rel + pos_embedding, B layout
  _Float16 h2B[ND * TB_STRIDE];    // h2, B layout
  float    gpe[NK * GPE_STRIDE];   // group_feat + pos_embedding, [k][c] col-major
  float    keyf[NC];
  float    gp[3 * NK];             // group_point
  int      idx[NK];
  float    kp[4];                  // key point xyz
};
struct BlockLds {
  GroupLds g[4];
  float pb2s[NC];
  float ab2rs[272];
  float ab1s[ND];
  float sc2s[ND];
  float sh2s[ND];
};

__global__ void __launch_bounds__(128)
agp_main(const float* __restrict__ vertices, const float* __restrict__ feature_map,
         const float* __restrict__ pw1, const float* __restrict__ pb1,
         const float* __restrict__ sc1, const float* __restrict__ sh1,
         const _Float16* __restrict__ pw2h, const float* __restrict__ pb2,
         const _Float16* __restrict__ aw1h, const float* __restrict__ ab1,
         const float* __restrict__ sc2, const float* __restrict__ sh2,
         const _Float16* __restrict__ aw2rh, const float* __restrict__ ab2r,
         const int* __restrict__ fps_idx, const int* __restrict__ knn_idx,
         float* __restrict__ out) {
  extern __shared__ __align__(16) char smem_raw[];
  BlockLds& BL = *(BlockLds*)smem_raw;
  const int tid = threadIdx.x;
  const int wid = tid >> 5, lane = tid & 31;
  const int lane15 = lane & 15, g = lane >> 4;
  const int grp = blockIdx.x * 4 + wid;
  const int b = grp >> 10, m = grp & (NM - 1);
  GroupLds& L = BL.g[wid];

  // block-shared constants
  for (int i = tid; i < NC; i += 128) BL.pb2s[i] = pb2[i];
  for (int i = tid; i < 272; i += 128) BL.ab2rs[i] = ab2r[i];
  if (tid < ND) { BL.ab1s[tid] = ab1[tid]; BL.sc2s[tid] = sc2[tid]; BL.sh2s[tid] = sh2[tid]; }

  const float* vb = vertices + (size_t)b * 3 * NP;
  const float* fm = feature_map + (size_t)b * NC * NP;
  const int kidx = fps_idx[b * NM + m];
  if (lane < NK) {
    const int n = knn_idx[(size_t)(b * NM + m) * NK + lane];
    L.idx[lane] = n;
#pragma unroll
    for (int c = 0; c < 3; ++c) L.gp[c * NK + lane] = vb[(size_t)c * NP + n];
  }
  if (lane < 3) L.kp[lane] = vb[(size_t)lane * NP + kidx];
  for (int c = lane; c < NC; c += 32) L.keyf[c] = fm[(size_t)c * NP + kidx];
  __syncthreads();

  // ---- stage 0: h = lrelu(bn1(pw1 @ pos_rel + pb1)), produced in B layout --
  // lane owns neighbor col lane15; halves = reduction rows d in table order.
  v16h hB[2];
#pragma unroll
  for (int dd = 0; dd < 2; ++dd) {
#pragma unroll
    for (int k = 0; k < NK; ++k) {
      const int d = 32 * dd + ((k < 8) ? (8 * g + k) : (8 + 8 * g + k));
      const float w0 = pw1[d * 3 + 0], w1 = pw1[d * 3 + 1], w2 = pw1[d * 3 + 2];
      const float bb = pb1[d], s = sc1[d], sh = sh1[d];
      float p0 = L.kp[0] - L.gp[0 * NK + lane15];
      float p1 = L.kp[1] - L.gp[1 * NK + lane15];
      float p2 = L.kp[2] - L.gp[2 * NK + lane15];
      float a = w0 * p0 + w1 * p1 + w2 * p2 + bb;
      a = a * s + sh;
      a = (a > 0.f) ? a : 0.2f * a;
      hB[dd][k] = (_Float16)a;
    }
  }

  // ---- stage 1: pos_embedding = pw2 @ h + pb2 (16 tiles x 2 WMMA) ---------
  // fused with group_feat gather; t = keyf - gf + pe -> LDS (B layout);
  // gpe = gf + pe -> LDS column-major (2x ds_store_b128 per tile).
  const int kcol = lane15;
  const int ncol = L.idx[kcol];
  for (int t = 0; t < 16; ++t) {
    const _Float16* arow = pw2h + (size_t)(16 * t + lane15) * ND;
    v8f acc = {};
    acc = wmma16(load_a(arow, g), hB[0], acc);
    acc = wmma16(load_a(arow + 32, g), hB[1], acc);
    const int c0 = 16 * t + 8 * g;                 // first C-layout row (channel)
    float4 kfa = *(const float4*)&L.keyf[c0];
    float4 kfb = *(const float4*)&L.keyf[c0 + 4];
    float4 pba = *(const float4*)&BL.pb2s[c0];
    float4 pbb = *(const float4*)&BL.pb2s[c0 + 4];
    const float kf[8] = {kfa.x, kfa.y, kfa.z, kfa.w, kfb.x, kfb.y, kfb.z, kfb.w};
    const float pbv[8] = {pba.x, pba.y, pba.z, pba.w, pbb.x, pbb.y, pbb.z, pbb.w};
    float gf[8];
#pragma unroll
    for (int r = 0; r < 8; ++r)
      gf[r] = fm[(size_t)(c0 + r) * NP + ncol];    // scattered gather (L2 hit)
    float gpev[8];
    _Float16 tv[8];
#pragma unroll
    for (int r = 0; r < 8; ++r) {
      float pe = acc[r] + pbv[r];
      gpev[r] = gf[r] + pe;
      tv[r] = (_Float16)(kf[r] - gf[r] + pe);
    }
    float* gdst = &L.gpe[kcol * GPE_STRIDE + c0];
    *(float4*)(gdst)     = make_float4(gpev[0], gpev[1], gpev[2], gpev[3]);
    *(float4*)(gdst + 4) = make_float4(gpev[4], gpev[5], gpev[6], gpev[7]);
#pragma unroll
    for (int r = 0; r < 8; ++r) L.tB[(c0 + r) * TB_STRIDE + kcol] = tv[r];
  }
  __syncthreads();

  // ---- stage 2: h2 = lrelu(bn2(aw1 @ t + ab1)) (4 tiles x 8 WMMA) ---------
  for (int t2 = 0; t2 < 4; ++t2) {
    v8f acc = {};
    const _Float16* arow = aw1h + (size_t)(16 * t2 + lane15) * NC;
    for (int kb = 0; kb < 8; ++kb) {
      v16h A  = load_a(arow + kb * 32, g);
      v16h Bv = load_b(L.tB + (size_t)(kb * 32 + 8 * g) * TB_STRIDE + lane15);
      acc = wmma16(A, Bv, acc);
    }
    const int d0 = 16 * t2 + 8 * g;
    float4 a1a = *(const float4*)&BL.ab1s[d0];
    float4 a1b = *(const float4*)&BL.ab1s[d0 + 4];
    float4 s2a = *(const float4*)&BL.sc2s[d0];
    float4 s2b = *(const float4*)&BL.sc2s[d0 + 4];
    float4 h2a = *(const float4*)&BL.sh2s[d0];
    float4 h2b = *(const float4*)&BL.sh2s[d0 + 4];
    const float a1[8] = {a1a.x, a1a.y, a1a.z, a1a.w, a1b.x, a1b.y, a1b.z, a1b.w};
    const float s2[8] = {s2a.x, s2a.y, s2a.z, s2a.w, s2b.x, s2b.y, s2b.z, s2b.w};
    const float h2s[8] = {h2a.x, h2a.y, h2a.z, h2a.w, h2b.x, h2b.y, h2b.z, h2b.w};
#pragma unroll
    for (int r = 0; r < 8; ++r) {
      float a = acc[r] + a1[r];
      a = a * s2[r] + h2s[r];
      a = (a > 0.f) ? a : 0.2f * a;
      L.h2B[(d0 + r) * TB_STRIDE + kcol] = (_Float16)a;
    }
  }
  __syncthreads();

  // ---- stage 3: logits = aw2r @ h2 + ab2r (17 tiles x 2 WMMA), softmax ----
  // over the 16 neighbors; pooled output = sum(e*gpe)/sum(e) (saves a pass).
  v16h B0 = load_b(L.h2B + (size_t)(8 * g) * TB_STRIDE + lane15);
  v16h B1 = load_b(L.h2B + (size_t)(32 + 8 * g) * TB_STRIDE + lane15);
  const size_t outb = (size_t)b * NOUT * NM + m;
  for (int t3 = 0; t3 < 17; ++t3) {
    const _Float16* arow = aw2rh + (size_t)(16 * t3 + lane15) * ND;
    v8f acc = {};
    acc = wmma16(load_a(arow, g), B0, acc);
    acc = wmma16(load_a(arow + 32, g), B1, acc);
    const int r0 = 16 * t3 + 8 * g;
    float4 b2a = *(const float4*)&BL.ab2rs[r0];
    float4 b2b = *(const float4*)&BL.ab2rs[r0 + 4];
    const float bb2[8] = {b2a.x, b2a.y, b2a.z, b2a.w, b2b.x, b2b.y, b2b.z, b2b.w};
    float gv[8];
    if (t3 < 16) {
      const float* gpr = &L.gpe[kcol * GPE_STRIDE + r0];
      float4 ga = *(const float4*)gpr;
      float4 gb = *(const float4*)(gpr + 4);
      gv[0] = ga.x; gv[1] = ga.y; gv[2] = ga.z; gv[3] = ga.w;
      gv[4] = gb.x; gv[5] = gb.y; gv[6] = gb.z; gv[7] = gb.w;
    } else {
#pragma unroll
      for (int r = 0; r < 8; ++r) {
        const int cx = 8 * g + r;
        gv[r] = (cx < 3) ? L.gp[cx * NK + kcol] : 0.f;
      }
    }
#pragma unroll
    for (int r = 0; r < 8; ++r) {
      float x = acc[r] + bb2[r];
      float mx = hredmax16(x);
      float e = __expf(x - mx);
      float den = hredsum16(e);
      float num = hredsum16(e * gv[r]);
      float red = num / den;
      if (t3 < 16) {
        if (lane15 == 0) out[outb + (size_t)(3 + r0 + r) * NM] = red;
      } else {
        const int cx = 8 * g + r;
        if (lane15 == 0 && cx < 3) out[outb + (size_t)cx * NM] = red;
      }
    }
  }
}

// ---------------------------------------------------------------------------
extern "C" void kernel_launch(void* const* d_in, const int* in_sizes, int n_in,
                              void* d_out, int out_size, void* d_ws, size_t ws_size,
                              hipStream_t stream) {
  (void)in_sizes; (void)n_in; (void)out_size; (void)ws_size;
  const float* vertices    = (const float*)d_in[0];
  const float* feature_map = (const float*)d_in[1];
  const float* pw1   = (const float*)d_in[2];
  const float* pb1   = (const float*)d_in[3];
  const float* bn1_g = (const float*)d_in[4];
  const float* bn1_b = (const float*)d_in[5];
  const float* bn1_m = (const float*)d_in[6];
  const float* bn1_v = (const float*)d_in[7];
  const float* pw2   = (const float*)d_in[8];
  const float* pb2   = (const float*)d_in[9];
  const float* aw1   = (const float*)d_in[10];
  const float* ab1   = (const float*)d_in[11];
  const float* bn2_g = (const float*)d_in[12];
  const float* bn2_b = (const float*)d_in[13];
  const float* bn2_m = (const float*)d_in[14];
  const float* bn2_v = (const float*)d_in[15];
  const float* aw2   = (const float*)d_in[16];
  const float* ab2   = (const float*)d_in[17];
  float* out = (float*)d_out;

  char* ws = (char*)d_ws;
  size_t off = 0;
  auto walloc = [&](size_t bytes) -> void* {
    void* p = ws + off;
    off = (off + bytes + 255) & ~(size_t)255;
    return p;
  };
  int*       fps_idx = (int*)walloc((size_t)NB * NM * sizeof(int));
  int*       knn_idx = (int*)walloc((size_t)NB * NM * NK * sizeof(int));
  _Float16*  pw2h    = (_Float16*)walloc((size_t)NC * ND * sizeof(_Float16));
  _Float16*  aw1h    = (_Float16*)walloc((size_t)ND * NC * sizeof(_Float16));
  _Float16*  aw2rh   = (_Float16*)walloc((size_t)272 * ND * sizeof(_Float16));
  float*     ab2r    = (float*)walloc(272 * sizeof(float));
  float*     sc1     = (float*)walloc(ND * sizeof(float));
  float*     sh1     = (float*)walloc(ND * sizeof(float));
  float*     sc2     = (float*)walloc(ND * sizeof(float));
  float*     sh2     = (float*)walloc(ND * sizeof(float));

  prep_kernel<<<dim3(32), dim3(256), 0, stream>>>(
      pw2, aw1, aw2, ab2, bn1_g, bn1_b, bn1_m, bn1_v, bn2_g, bn2_b, bn2_m, bn2_v,
      pw2h, aw1h, aw2rh, ab2r, sc1, sh1, sc2, sh2);

  fps_kernel<<<dim3(NB), dim3(1024), 3 * NP * sizeof(float), stream>>>(vertices, fps_idx);

  knn_kernel<<<dim3((NB * NM) / 256), dim3(256), 3 * 1024 * sizeof(float), stream>>>(
      vertices, fps_idx, knn_idx);

  agp_main<<<dim3((NB * NM) / 4), dim3(128), sizeof(BlockLds), stream>>>(
      vertices, feature_map, pw1, pb1, sc1, sh1, pw2h, pb2, aw1h, ab1, sc2, sh2,
      aw2rh, ab2r, fps_idx, knn_idx, out);
}
